// GCNmfConv_74088185856113
// MI455X (gfx1250) — compile-verified, matrix operand order checked by reading, not run on hardware
//
#include <hip/hip_runtime.h>
#include <hip/hip_bf16.h>
#include <math.h>

// ---------------------------------------------------------------------------
// GCNmf forward on gfx1250 (MI455X).
// N=10000 nodes, F=256 in-features, O=128 out-features, K=5 GMM components,
// E=160000 edges.
// ---------------------------------------------------------------------------

#define NNODE 10000
#define NF    256
#define NO    128
#define NK    5
#define NE    160000

typedef __attribute__((ext_vector_type(16))) __bf16 v16bf;
typedef __attribute__((ext_vector_type(8)))  float  v8f;

// ---------------------------------------------------------------------------
// Kernel 1: pack W and W*W into per-lane WMMA B-fragment layout (bf16), and
// compute variances = exp(logvars).
// Fragment layout (matches 16-bit A/B VGPR striping in CDNA5 ISA):
//   fragment index (kt, j): kt = K-tile (32 features), j = N-tile (16 outputs)
//   lane l: column o = j*16 + (l & 15); khalf = l >> 4
//   element e (0..15): k = (e<8) ? khalf*8+e : 16 + khalf*8 + (e-8)
//   f = kt*32 + k
// ---------------------------------------------------------------------------
__global__ void __launch_bounds__(256)
pack_kernel(const float* __restrict__ weight, const float* __restrict__ logvars,
            __bf16* __restrict__ wpk, __bf16* __restrict__ w2pk,
            float* __restrict__ vars) {
  const int idx = blockIdx.x * 256 + threadIdx.x;   // 0 .. 32767
  const int e     = idx & 15;
  const int lane  = (idx >> 4) & 31;
  const int j     = (idx >> 9) & 7;
  const int kt    = idx >> 12;
  const int khalf = lane >> 4;
  const int kk    = (e < 8) ? (khalf * 8 + e) : (16 + khalf * 8 + (e - 8));
  const int f     = kt * 32 + kk;
  const int o     = j * 16 + (lane & 15);
  const float wv  = weight[f * NO + o];
  wpk[idx]  = (__bf16)wv;
  w2pk[idx] = (__bf16)(wv * wv);
  if (idx < NK * NF) vars[idx] = expf(logvars[idx]);
}

// ---------------------------------------------------------------------------
// Kernel 2: responsibilities gamma[k][n].
// One wave32 per node; lane strides over features; shuffle-xor reduction.
// Constant terms of log_n cancel in the softmax over k.
// ---------------------------------------------------------------------------
__global__ void __launch_bounds__(256)
gamma_kernel(const float* __restrict__ x, const unsigned char* __restrict__ mask,
             const float* __restrict__ means, const float* __restrict__ vars,
             const float* __restrict__ logp, float* __restrict__ gamma, int n) {
  const int wave = threadIdx.x >> 5;
  const int lane = threadIdx.x & 31;
  const int node = blockIdx.x * 8 + wave;
  if (node >= n) return;

  float s[NK] = {0.f, 0.f, 0.f, 0.f, 0.f};
  for (int f = lane; f < NF; f += 32) {
    const float xv = x[(size_t)node * NF + f];
    const bool miss = mask[(size_t)node * NF + f] != 0;
    if (!miss) {
#pragma unroll
      for (int k = 0; k < NK; ++k) {
        const float d = xv - means[k * NF + f];
        s[k] += d * d / vars[k * NF + f];
      }
    }
  }
#pragma unroll
  for (int k = 0; k < NK; ++k)
#pragma unroll
    for (int off = 16; off > 0; off >>= 1)
      s[k] += __shfl_xor(s[k], off, 32);

  if (lane == 0) {
    float t[NK], mx = -3.0e38f;
#pragma unroll
    for (int k = 0; k < NK; ++k) { t[k] = logp[k] - 0.5f * s[k]; mx = fmaxf(mx, t[k]); }
    float sum = 0.f;
#pragma unroll
    for (int k = 0; k < NK; ++k) { t[k] = expf(t[k] - mx); sum += t[k]; }
    const float inv = 1.0f / sum;
#pragma unroll
    for (int k = 0; k < NK; ++k) gamma[(size_t)k * n + node] = t[k] * inv;
  }
}

// ---------------------------------------------------------------------------
// Kernel 3: zero conv accumulators (must run every launch; atomics accumulate).
// ---------------------------------------------------------------------------
__global__ void __launch_bounds__(256)
zero_kernel(float4* __restrict__ p, int n4) {
  const int i = blockIdx.x * 256 + threadIdx.x;
  if (i < n4) p[i] = make_float4(0.f, 0.f, 0.f, 0.f);
}

// ---------------------------------------------------------------------------
// Kernel 4: fused masked-impute GEMM via WMMA bf16.
//   tx[k,n,o] = (mask ? means[k] : x) @ W + bias
//   tc[k,n,o] = (mask ? vars[k]  : 0) @ (W*W)
// Block = 256 threads = 8 waves; wave w owns rows [tile*128 + w*16, +16),
// full O=128 => 8 col-tiles x 2 outputs = 16 v8f accumulators per wave.
// A fragments are built in-register (mask-select fused into the load path).
// ---------------------------------------------------------------------------
__global__ void __launch_bounds__(256)
gemm_kernel(const float* __restrict__ x, const unsigned char* __restrict__ mask,
            const float* __restrict__ means, const float* __restrict__ vars,
            const __bf16* __restrict__ wpk, const __bf16* __restrict__ w2pk,
            const float* __restrict__ bias,
            float* __restrict__ tx, float* __restrict__ tc, int n) {
  const int k      = blockIdx.y;
  const int wave   = threadIdx.x >> 5;
  const int lane   = threadIdx.x & 31;
  const int lrow   = lane & 15;
  const int khalf  = lane >> 4;
  const int rowBase = blockIdx.x * 128 + wave * 16;
  int m = rowBase + lrow;
  if (m >= n) m = n - 1;              // clamp: keep EXEC all-ones for WMMA

  v8f accX[8], accC[8];
#pragma unroll
  for (int j = 0; j < 8; ++j) { accX[j] = (v8f)0.0f; accC[j] = (v8f)0.0f; }

  const v16bf* wfrag  = reinterpret_cast<const v16bf*>(wpk);
  const v16bf* w2frag = reinterpret_cast<const v16bf*>(w2pk);

#pragma unroll
  for (int kt = 0; kt < 8; ++kt) {
    const int f0 = kt * 32 + khalf * 8;   // first 8-feature run for this lane
    const int f1 = f0 + 16;               // second run

    float xa[16], mm[16], vv[16];
    unsigned char mb[16];
#pragma unroll
    for (int i = 0; i < 8; ++i) {
      xa[i]     = x[(size_t)m * NF + f0 + i];
      xa[8 + i] = x[(size_t)m * NF + f1 + i];
      mb[i]     = mask[(size_t)m * NF + f0 + i];
      mb[8 + i] = mask[(size_t)m * NF + f1 + i];
      mm[i]     = means[k * NF + f0 + i];
      mm[8 + i] = means[k * NF + f1 + i];
      vv[i]     = vars[k * NF + f0 + i];
      vv[8 + i] = vars[k * NF + f1 + i];
    }

    v16bf af, ac;
#pragma unroll
    for (int e = 0; e < 16; ++e) {
      const bool miss = mb[e] != 0;
      af[e] = (__bf16)(miss ? mm[e] : xa[e]);
      ac[e] = (__bf16)(miss ? vv[e] : 0.0f);
    }

#pragma unroll
    for (int j = 0; j < 8; ++j) {
      const v16bf bw  = wfrag [(size_t)(kt * 8 + j) * 32 + lane];
      const v16bf bw2 = w2frag[(size_t)(kt * 8 + j) * 32 + lane];
      accX[j] = __builtin_amdgcn_wmma_f32_16x16x32_bf16(
          false, af, false, bw,  (short)0, accX[j], false, false);
      accC[j] = __builtin_amdgcn_wmma_f32_16x16x32_bf16(
          false, ac, false, bw2, (short)0, accC[j], false, false);
    }
  }

  // Writeback: C/D layout -> VGPR r holds row (khalf*8 + r), col = lane&15.
#pragma unroll
  for (int j = 0; j < 8; ++j) {
    const int col = j * 16 + lrow;
    const float b = bias[col];
#pragma unroll
    for (int r = 0; r < 8; ++r) {
      const int row = rowBase + khalf * 8 + r;
      if (row < n) {
        const size_t off = ((size_t)k * n + row) * NO + col;
        tx[off] = accX[j][r] + b;
        tc[off] = accC[j][r];
      }
    }
  }
}

// ---------------------------------------------------------------------------
// Kernel 5: SPMM scatter. conv_x[k,row,:] += val * tx[k,col,:]
//           conv_c[k,row,:] += val^2 * tc[k,col,:]
// 2 edges per 256-thread block; f32 global atomics (targets fit in 192MB L2).
// ---------------------------------------------------------------------------
__global__ void __launch_bounds__(256)
spmm_kernel(const int* __restrict__ er, const int* __restrict__ ec,
            const float* __restrict__ ev,
            const float* __restrict__ tx, const float* __restrict__ tc,
            float* __restrict__ cx, float* __restrict__ cc, int nE, int n) {
  const int t = blockIdx.x * 256 + threadIdx.x;
  const int e = t >> 7;
  const int o = t & 127;
  if (e >= nE) return;
  const int r = er[e];
  const int c = ec[e];
  const float v  = ev[e];
  const float v2 = v * v;
  const size_t rb = (size_t)r * NO + o;
  const size_t cb = (size_t)c * NO + o;
  const size_t ks = (size_t)n * NO;
#pragma unroll
  for (int k = 0; k < NK; ++k) {
    atomicAdd(cx + (size_t)k * ks + rb, v  * tx[(size_t)k * ks + cb]);
    atomicAdd(cc + (size_t)k * ks + rb, v2 * tc[(size_t)k * ks + cb]);
  }
}

// ---------------------------------------------------------------------------
// Kernel 6: out[n,o] = sum_k gamma[k,n] * ex_relu(conv_x[k,n,o], conv_c[k,n,o])
// ---------------------------------------------------------------------------
__global__ void __launch_bounds__(256)
final_kernel(const float* __restrict__ cx, const float* __restrict__ cc,
             const float* __restrict__ gamma, float* __restrict__ out, int n) {
  const int idx = blockIdx.x * 256 + threadIdx.x;
  if (idx >= n * NO) return;
  const int node = idx / NO;
  const size_t ks = (size_t)n * NO;
  const float inv_s2pi = 0.3989422804014327f;   // 1/sqrt(2*pi)
  const float inv_s2   = 0.7071067811865476f;   // 1/sqrt(2)
  float acc = 0.f;
#pragma unroll
  for (int k = 0; k < NK; ++k) {
    const float mu = cx[(size_t)k * ks + idx];
    const float sg = cc[(size_t)k * ks + idx];
    float ex;
    if (sg == 0.0f) {
      ex = fmaxf(mu, 0.0f);
    } else {
      const float sq = sqrtf(sg);
      const float w  = mu / sq;
      ex = sq * (expf(-0.5f * w * w) * inv_s2pi +
                 0.5f * w * (1.0f + erff(w * inv_s2)));
    }
    acc += gamma[(size_t)k * n + node] * ex;
  }
  out[idx] = acc;
}

// ---------------------------------------------------------------------------
// Host launcher
// ---------------------------------------------------------------------------
extern "C" void kernel_launch(void* const* d_in, const int* in_sizes, int n_in,
                              void* d_out, int out_size, void* d_ws, size_t ws_size,
                              hipStream_t stream) {
  const float*         x_vals  = (const float*)d_in[0];
  const unsigned char* x_mask  = (const unsigned char*)d_in[1];
  const int*           e_row   = (const int*)d_in[2];
  const int*           e_col   = (const int*)d_in[3];
  const float*         e_val   = (const float*)d_in[4];
  const float*         logp    = (const float*)d_in[5];
  const float*         means   = (const float*)d_in[6];
  const float*         logvars = (const float*)d_in[7];
  const float*         weight  = (const float*)d_in[8];
  const float*         bias    = (const float*)d_in[9];
  float* out = (float*)d_out;

  const int n = NNODE, nE = NE;

  // workspace carve-up (256B aligned)
  char* ws = (char*)d_ws;
  size_t p = 0;
  auto alloc = [&](size_t bytes) -> size_t {
    size_t o = p;
    p = (p + bytes + 255) & ~(size_t)255;
    return o;
  };
  const size_t off_gamma = alloc((size_t)NK * n * sizeof(float));
  const size_t off_vars  = alloc((size_t)NK * NF * sizeof(float));
  const size_t off_wpk   = alloc((size_t)NF * NO * sizeof(__bf16));
  const size_t off_w2pk  = alloc((size_t)NF * NO * sizeof(__bf16));
  const size_t off_tx    = alloc((size_t)NK * n * NO * sizeof(float));
  const size_t off_tc    = alloc((size_t)NK * n * NO * sizeof(float));
  const size_t off_cx    = alloc((size_t)NK * n * NO * sizeof(float));
  const size_t off_cc    = alloc((size_t)NK * n * NO * sizeof(float));
  (void)ws_size; (void)in_sizes; (void)n_in; (void)out_size;

  float*  g_gamma = (float*)(ws + off_gamma);
  float*  g_vars  = (float*)(ws + off_vars);
  __bf16* g_wpk   = (__bf16*)(ws + off_wpk);
  __bf16* g_w2pk  = (__bf16*)(ws + off_w2pk);
  float*  g_tx    = (float*)(ws + off_tx);
  float*  g_tc    = (float*)(ws + off_tc);
  float*  g_cx    = (float*)(ws + off_cx);
  float*  g_cc    = (float*)(ws + off_cc);

  // 1) pack W / W^2 fragments + variances
  pack_kernel<<<(NF * NO) / 256, 256, 0, stream>>>(weight, logvars, g_wpk, g_w2pk, g_vars);

  // 2) responsibilities
  gamma_kernel<<<(n + 7) / 8, 256, 0, stream>>>(x_vals, x_mask, means, g_vars,
                                                logp, g_gamma, n);

  // 3) zero conv accumulators (cx and cc are contiguous)
  {
    const int n4 = (2 * NK * n * NO) / 4;
    zero_kernel<<<(n4 + 255) / 256, 256, 0, stream>>>((float4*)g_cx, n4);
  }

  // 4) fused impute + GEMM (WMMA bf16), per-k
  {
    dim3 grid((n + 127) / 128, NK, 1);
    gemm_kernel<<<grid, 256, 0, stream>>>(x_vals, x_mask, means, g_vars,
                                          g_wpk, g_w2pk, bias, g_tx, g_tc, n);
  }

  // 5) SPMM scatter with f32 atomics
  spmm_kernel<<<(nE * NO) / 256, 256, 0, stream>>>(e_row, e_col, e_val,
                                                   g_tx, g_tc, g_cx, g_cc, nE, n);

  // 6) ex_relu * gamma reduction over k
  final_kernel<<<(n * NO + 255) / 256, 256, 0, stream>>>(g_cx, g_cc, g_gamma, out, n);
}
